// GCN_24318104830697
// MI455X (gfx1250) — compile-verified, matrix operand order checked by reading, not run on hardware
//
#include <hip/hip_runtime.h>
#include <hip/hip_bf16.h>
#include <math.h>

// ---------------------------------------------------------------------------
// GCN (4x GCNConv, D=64) for MI455X / gfx1250.
//   - per-edge norm precomputed once, edge indices narrowed to i32 once
//   - dense H@W done with CDNA5 WMMA (exact f32 16x16x4; bf16 fallback)
//   - scatter via native f32 global atomics (L2-resident: 25.6MB << 192MB L2)
//   - ELU fused into the next layer's GEMM operand load (branchless v_exp_f32),
//     selected at compile time via template so the K-loop is straight-line code
// ---------------------------------------------------------------------------

typedef __attribute__((ext_vector_type(2)))  float v2f;
typedef __attribute__((ext_vector_type(8)))  float v8f;

#if defined(__has_builtin)
#if __has_builtin(__builtin_amdgcn_wmma_f32_16x16x4_f32)
#define GCN_USE_WMMA_F32 1
#endif
#endif

#ifndef GCN_USE_WMMA_F32
typedef __attribute__((ext_vector_type(16))) __bf16 v16bf;
#endif

// Branchless ELU: v_min + v_exp_f32 + v_add + v_cndmask, no EXEC manipulation
// (keeps EXEC all-ones through the WMMA stream as the ISA requires).
__device__ __forceinline__ float eluf(float x) {
  float e = __expf(fminf(x, 0.0f)) - 1.0f;
  return x > 0.0f ? x : e;
}

// ---------------- degree / norm prep ----------------

__global__ void k_init_deg(int* __restrict__ cnt, int N) {
  int i = blockIdx.x * blockDim.x + threadIdx.x;
  if (i < N) cnt[i] = 1;  // self loop
}

__global__ void k_edge_prep(const long long* __restrict__ ei,
                            int* __restrict__ src32, int* __restrict__ dst32,
                            int* __restrict__ cnt, int E) {
  int e = blockIdx.x * blockDim.x + threadIdx.x;
  if (e >= E) return;
  int s = (int)ei[e];
  int d = (int)ei[(size_t)E + e];
  src32[e] = s;
  dst32[e] = d;
  atomicAdd(&cnt[d], 1);
}

__global__ void k_finish_deg(float* __restrict__ dis, int N) {
  int i = blockIdx.x * blockDim.x + threadIdx.x;
  if (i >= N) return;
  int c = ((const int*)dis)[i];      // reuse buffer: counts -> rsqrt(deg)
  dis[i] = rsqrtf((float)c);         // c >= 1 always (self loop)
}

__global__ void k_edge_norm(const int* __restrict__ src32,
                            const int* __restrict__ dst32,
                            const float* __restrict__ dis,
                            float* __restrict__ norm, int E) {
  int e = blockIdx.x * blockDim.x + threadIdx.x;
  if (e >= E) return;
  norm[e] = dis[src32[e]] * dis[dst32[e]];
}

// ---------------- dense H @ W via WMMA ----------------
// Block = 256 threads = 8 waves; each wave computes a 16x64 output stripe.
// APPLY_ELU is a compile-time parameter: layer 1 gets a pure load/WMMA stream,
// layers 2-4 get branchless ELU inlined with no per-step scalar checks.

template <bool APPLY_ELU>
__global__ __launch_bounds__(256)
void k_gemm64(const float* __restrict__ H, const float* __restrict__ W,
              float* __restrict__ T, int N) {
  __shared__ float Wt[64 * 64];  // W transposed: Wt[n*64 + k]
  int tid = threadIdx.x;
  #pragma unroll
  for (int i = 0; i < 16; ++i) {
    int idx = tid + i * 256;           // flat over 4096
    int k = idx >> 6, n = idx & 63;
    Wt[n * 64 + k] = W[idx];
  }
  __syncthreads();

  int wave = tid >> 5;
  int lane = tid & 31;
  int halfsel = lane >> 4;             // 0: lanes 0-15, 1: lanes 16-31
  int l15 = lane & 15;
  int r0 = blockIdx.x * 128 + wave * 16;

  int arow = r0 + l15;
  if (arow >= N) arow = N - 1;         // clamp (stores are guarded)
  const float* Arow = H + (size_t)arow * 64;

  v8f c0 = {}, c1 = {}, c2 = {}, c3 = {};

#ifdef GCN_USE_WMMA_F32
  #pragma unroll
  for (int k0 = 0; k0 < 64; k0 += 4) {
    int ka = k0 + halfsel * 2;         // 32-bit A layout: lanes 16-31 hold K+2
    float ax = Arow[ka], ay = Arow[ka + 1];
    if (APPLY_ELU) { ax = eluf(ax); ay = eluf(ay); }
    v2f a; a.x = ax; a.y = ay;
    v2f b0, b1, b2, b3;
    b0.x = Wt[(0 * 16 + l15) * 64 + ka]; b0.y = Wt[(0 * 16 + l15) * 64 + ka + 1];
    b1.x = Wt[(1 * 16 + l15) * 64 + ka]; b1.y = Wt[(1 * 16 + l15) * 64 + ka + 1];
    b2.x = Wt[(2 * 16 + l15) * 64 + ka]; b2.y = Wt[(2 * 16 + l15) * 64 + ka + 1];
    b3.x = Wt[(3 * 16 + l15) * 64 + ka]; b3.y = Wt[(3 * 16 + l15) * 64 + ka + 1];
    c0 = __builtin_amdgcn_wmma_f32_16x16x4_f32(false, a, false, b0, (short)0, c0, false, false);
    c1 = __builtin_amdgcn_wmma_f32_16x16x4_f32(false, a, false, b1, (short)0, c1, false, false);
    c2 = __builtin_amdgcn_wmma_f32_16x16x4_f32(false, a, false, b2, (short)0, c2, false, false);
    c3 = __builtin_amdgcn_wmma_f32_16x16x4_f32(false, a, false, b3, (short)0, c3, false, false);
  }
#else
  // bf16 fallback: K=64 as two 16x16x32 steps (codegen-confirmed builtin)
  int koff8 = halfsel * 8;
  #pragma unroll
  for (int kb = 0; kb < 64; kb += 32) {
    v16bf a;
    #pragma unroll
    for (int e2 = 0; e2 < 16; ++e2) {
      int K = kb + koff8 + (e2 & 7) + ((e2 >= 8) ? 16 : 0);
      float v = Arow[K];
      if (APPLY_ELU) v = eluf(v);
      a[e2] = (__bf16)v;
    }
    v16bf b0, b1, b2, b3;
    #pragma unroll
    for (int e2 = 0; e2 < 16; ++e2) {
      int K = kb + koff8 + (e2 & 7) + ((e2 >= 8) ? 16 : 0);
      b0[e2] = (__bf16)Wt[(0 * 16 + l15) * 64 + K];
      b1[e2] = (__bf16)Wt[(1 * 16 + l15) * 64 + K];
      b2[e2] = (__bf16)Wt[(2 * 16 + l15) * 64 + K];
      b3[e2] = (__bf16)Wt[(3 * 16 + l15) * 64 + K];
    }
    c0 = __builtin_amdgcn_wmma_f32_16x16x32_bf16(false, a, false, b0, (short)0, c0, false, false);
    c1 = __builtin_amdgcn_wmma_f32_16x16x32_bf16(false, a, false, b1, (short)0, c1, false, false);
    c2 = __builtin_amdgcn_wmma_f32_16x16x32_bf16(false, a, false, b2, (short)0, c2, false, false);
    c3 = __builtin_amdgcn_wmma_f32_16x16x32_bf16(false, a, false, b3, (short)0, c3, false, false);
  }
#endif

  // C/D layout: VGPR r -> row M=r (lanes 0-15) / M=8+r (lanes 16-31), col = lane&15
  #pragma unroll
  for (int r = 0; r < 8; ++r) {
    int row = r0 + halfsel * 8 + r;
    if (row < N) {
      float* dst = T + (size_t)row * 64 + l15;
      dst[0]  = c0[r];
      dst[16] = c1[r];
      dst[32] = c2[r];
      dst[48] = c3[r];
    }
  }
}

// ---------------- aggregation ----------------

// Out = T * dis^2 (self loop) + bias
__global__ void k_agg_init(const float* __restrict__ T,
                           const float* __restrict__ dis,
                           const float* __restrict__ bias,
                           float* __restrict__ Out, int N) {
  int i = blockIdx.x * blockDim.x + threadIdx.x;
  if (i >= N * 64) return;
  int node = i >> 6, f = i & 63;
  float d = dis[node];
  Out[i] = T[i] * d * d + bias[f];
}

// one wave per edge: 32 lanes x 2 feats, native f32 atomics (L2-resident)
__global__ void k_scatter(const float* __restrict__ T,
                          const int* __restrict__ src32,
                          const int* __restrict__ dst32,
                          const float* __restrict__ norm,
                          float* __restrict__ Out, int E) {
  int gid = blockIdx.x * blockDim.x + threadIdx.x;
  int e = gid >> 5;
  int lane = gid & 31;
  if (e >= E) return;
  int s = src32[e], d = dst32[e];
  float w = norm[e];
  const float* ts = T + (size_t)s * 64;
  float* od = Out + (size_t)d * 64;
  unsafeAtomicAdd(&od[lane],      ts[lane]      * w);
  unsafeAtomicAdd(&od[lane + 32], ts[lane + 32] * w);
}

// ---------------- host orchestration ----------------

extern "C" void kernel_launch(void* const* d_in, const int* in_sizes, int n_in,
                              void* d_out, int out_size, void* d_ws, size_t ws_size,
                              hipStream_t stream) {
  const float*     x  = (const float*)d_in[0];
  const long long* ei = (const long long*)d_in[1];
  // d_in[2] = batch (unused)
  const float* Wl[4] = {(const float*)d_in[3], (const float*)d_in[5],
                        (const float*)d_in[7], (const float*)d_in[9]};
  const float* bl[4] = {(const float*)d_in[4], (const float*)d_in[6],
                        (const float*)d_in[8], (const float*)d_in[10]};

  const int N = in_sizes[0] / 64;
  const int E = in_sizes[1] / 2;

  // workspace carve-up (floats/ints, all 4B; base is 256B aligned by harness)
  float* dis   = (float*)d_ws;               // N   (int counts first, then rsqrt)
  int*   src32 = (int*)(dis + N);            // E
  int*   dst32 = src32 + E;                  // E
  float* nrm   = (float*)(dst32 + E);        // E
  float* T     = nrm + E;                    // N*64
  float* P     = T + (size_t)N * 64;         // N*64
  float* Q     = P + (size_t)N * 64;         // N*64

  const int TB = 256;
  dim3 blk(TB);
  dim3 gN((N + TB - 1) / TB);
  dim3 gE((E + TB - 1) / TB);
  dim3 gNF(((size_t)N * 64 + TB - 1) / TB);
  dim3 gSc(((size_t)E * 32 + TB - 1) / TB);
  dim3 gGemm((N + 127) / 128);

  k_init_deg<<<gN, blk, 0, stream>>>((int*)dis, N);
  k_edge_prep<<<gE, blk, 0, stream>>>(ei, src32, dst32, (int*)dis, E);
  k_finish_deg<<<gN, blk, 0, stream>>>(dis, N);
  k_edge_norm<<<gE, blk, 0, stream>>>(src32, dst32, dis, nrm, E);

  float* out = (float*)d_out;
  const float* hin[4] = {x, P, Q, P};
  float*       hout[4] = {P, Q, P, out};
  for (int l = 0; l < 4; ++l) {
    if (l == 0) {
      k_gemm64<false><<<gGemm, blk, 0, stream>>>(hin[l], Wl[l], T, N);
    } else {
      k_gemm64<true><<<gGemm, blk, 0, stream>>>(hin[l], Wl[l], T, N);
    }
    k_agg_init<<<gNF, blk, 0, stream>>>(T, dis, bl[l], hout[l], N);
    k_scatter<<<gSc, blk, 0, stream>>>(T, src32, dst32, nrm, hout[l], E);
  }
}
